// R2IRCrossBlock_3487513445073
// MI455X (gfx1250) — matrix operand order repaired and checked by reading, not compiled, not run on hardware
//
#include <hip/hip_runtime.h>
#include <hip/hip_bf16.h>

// ---------------- types ----------------
typedef __attribute__((ext_vector_type(16))) __bf16 bf16x16;
typedef __attribute__((ext_vector_type(8)))  __bf16 bf16x8;
typedef __attribute__((ext_vector_type(8)))  float  floatx8;
typedef __attribute__((ext_vector_type(4)))  unsigned int u32x4;
typedef __attribute__((ext_vector_type(8)))  int i32x8;
typedef __attribute__((ext_vector_type(4)))  int i32x4;

#define CEMB 384
#define NQTOK 4096
#define NKTOK 1024
#define NHEAD 8
#define DHEAD 48
#define FFDIM 1536

// TDM availability (device pass only; host pass falls back to plain C++ copy)
#if defined(__HIP_DEVICE_COMPILE__) && defined(__has_builtin)
#if __has_builtin(__builtin_amdgcn_tensor_load_to_lds)
#define HAVE_TDM 1
#endif
#endif
#ifndef HAVE_TDM
#define HAVE_TDM 0
#endif

// ---------------- helpers ----------------
__device__ __forceinline__ unsigned short f2bf(float x) {
    unsigned int u = __float_as_uint(x);
    unsigned int r = u + 0x7FFFu + ((u >> 16) & 1u);   // round-to-nearest-even
    return (unsigned short)(r >> 16);
}
__device__ __forceinline__ float bf2f(unsigned short u) {
    return __uint_as_float(((unsigned int)u) << 16);
}

// A fragment: 16x32 bf16, row-major tile in LDS; lane(0..15)=row M, lanes 16..31 same rows.
// element e<8  -> K = hi*8 + e ; element e>=8 -> K = 16 + hi*8 + (e-8)
__device__ __forceinline__ bf16x16 load_a_frag(const unsigned short* rowp, int hi) {
    bf16x8 lo = *(const bf16x8*)(rowp + hi * 8);
    bf16x8 hv = *(const bf16x8*)(rowp + 16 + hi * 8);
    bf16x16 a;
#pragma unroll
    for (int e = 0; e < 8; ++e) { a[e] = lo[e]; a[e + 8] = hv[e]; }
    return a;
}
// B fragment: 32x16 bf16; LDS holds Bt[n][k] (one row per output column n).
// lane n = lane&15, element e<8 -> K = hi*8 + e ; element e>=8 -> K = 16 + hi*8 + (e-8)
__device__ __forceinline__ bf16x16 load_b_frag(const unsigned short* rowp, int hi) {
    bf16x8 lo = *(const bf16x8*)(rowp + hi * 8);
    bf16x8 hv = *(const bf16x8*)(rowp + 16 + hi * 8);
    bf16x16 b;
#pragma unroll
    for (int e = 0; e < 8; ++e) { b[e] = lo[e]; b[e + 8] = hv[e]; }
    return b;
}

// ---------------- small kernels ----------------
__global__ void f32_to_bf16_kernel(const float* __restrict__ in,
                                   unsigned short* __restrict__ out, int n) {
    int i = blockIdx.x * 256 + threadIdx.x;
    if (i < n) out[i] = f2bf(in[i]);
}

// gx[b*C+c] = sqrt( sum_{hw} x^2 )
__global__ void grn_reduce_kernel(const float* __restrict__ x, float* __restrict__ gx, int HW) {
    __shared__ float red[256];
    int bc = blockIdx.x, tid = threadIdx.x;
    const float* p = x + (size_t)bc * HW;
    float s = 0.f;
    for (int i = tid; i < HW; i += 256) { float v = p[i]; s += v * v; }
    red[tid] = s; __syncthreads();
    for (int st = 128; st > 0; st >>= 1) { if (tid < st) red[tid] += red[tid + st]; __syncthreads(); }
    if (tid == 0) gx[bc] = sqrtf(red[0]);
}

// alpha[b*C+c] = (1+gamma[c]) * gx/(mean_c(gx)+1e-6) + 1    (GRN folded to affine)
__global__ void grn_final_kernel(const float* __restrict__ gx, const float* __restrict__ gamma,
                                 float* __restrict__ alpha) {
    __shared__ float sg[CEMB];
    __shared__ float smean;
    int b = blockIdx.x, tid = threadIdx.x;
    float g = gx[b * CEMB + tid];
    sg[tid] = g; __syncthreads();
    if (tid == 0) {
        float s = 0.f;
        for (int i = 0; i < CEMB; ++i) s += sg[i];
        smean = s / (float)CEMB;
    }
    __syncthreads();
    float nx = g / (smean + 1e-6f);
    alpha[b * CEMB + tid] = (1.0f + gamma[tid]) * nx + 1.0f;
}

// per-token layernorm over channels (channel-major) + elu(y)+1
__global__ void ln_elu_kernel(const float* __restrict__ pre, const float* __restrict__ w,
                              const float* __restrict__ bg, float* __restrict__ out, int Ntok) {
    int t = blockIdx.x * 256 + threadIdx.x;
    int b = t / Ntok, n = t % Ntok;
    const float* p = pre + (size_t)b * CEMB * Ntok + n;
    float s = 0.f, s2 = 0.f;
    for (int f = 0; f < CEMB; ++f) { float v = p[(size_t)f * Ntok]; s += v; s2 += v * v; }
    float mu = s / (float)CEMB;
    float var = s2 / (float)CEMB - mu * mu;
    float inv = rsqrtf(var + 1e-5f);
    float* o = out + (size_t)b * CEMB * Ntok + n;
    for (int f = 0; f < CEMB; ++f) {
        float v = p[(size_t)f * Ntok];
        float y = (v - mu) * inv * w[f] + bg[f];
        o[(size_t)f * Ntok] = (y > 0.f) ? (y + 1.f) : __expf(y);
    }
}

// ksum[b*C+c] = sum_n k[b][c][n]
__global__ void ksum_kernel(const float* __restrict__ kpost, float* __restrict__ ks, int Nk) {
    __shared__ float red[256];
    int bc = blockIdx.x, tid = threadIdx.x;
    const float* p = kpost + (size_t)bc * Nk;
    float s = 0.f;
    for (int i = tid; i < Nk; i += 256) s += p[i];
    red[tid] = s; __syncthreads();
    for (int st = 128; st > 0; st >>= 1) { if (tid < st) red[tid] += red[tid + st]; __syncthreads(); }
    if (tid == 0) ks[bc] = red[0];
}

// kv[b][h][d][e] = sum_n K[h*48+d][n] * V[h*48+e][n]  via WMMA; 1 WG per (b,h)
__global__ __launch_bounds__(256) void kv_outer_kernel(const float* __restrict__ kpost,
                                                       const float* __restrict__ vbuf,
                                                       float* __restrict__ kv) {
    __shared__ __align__(16) unsigned short sK[48 * 40];
    __shared__ __align__(16) unsigned short sV[48 * 40];
    int bh = blockIdx.x;
    int b = bh >> 3, h = bh & 7;
    const float* Kp = kpost + ((size_t)b * CEMB + h * DHEAD) * NKTOK;
    const float* Vp = vbuf + ((size_t)b * CEMB + h * DHEAD) * NKTOK;
    int tid = threadIdx.x, lane = tid & 31, wave = tid >> 5;
    int mrow = lane & 15, hi = lane >> 4;
    int i0 = wave / 3, j0 = wave % 3;          // wave -> output tile (d-tile, e-tile), covers 8 of 9
    bool extra = (wave == 0);                  // wave 0 also does tile (2,2)
    floatx8 acc = {};
    floatx8 accx = {};
    for (int n0 = 0; n0 < NKTOK; n0 += 32) {
        __syncthreads();
        // 48*32 = 1536 elements, exactly 6 per thread -> constant-trip, branch-free staging
#pragma unroll
        for (int it = 0; it < 6; ++it) {
            int idx = tid + it * 256;
            int r = idx >> 5, c = idx & 31;
            sK[r * 40 + c] = f2bf(Kp[(size_t)r * NKTOK + n0 + c]);
            sV[r * 40 + c] = f2bf(Vp[(size_t)r * NKTOK + n0 + c]);
        }
        __syncthreads();
        bf16x16 a = load_a_frag(&sK[(i0 * 16 + mrow) * 40], hi);
        bf16x16 bb = load_b_frag(&sV[(j0 * 16 + mrow) * 40], hi);
        acc = __builtin_amdgcn_wmma_f32_16x16x32_bf16(false, a, false, bb, (short)0, acc, false, false);
        if (extra) {
            bf16x16 a2 = load_a_frag(&sK[(32 + mrow) * 40], hi);
            bf16x16 b2 = load_b_frag(&sV[(32 + mrow) * 40], hi);
            accx = __builtin_amdgcn_wmma_f32_16x16x32_bf16(false, a2, false, b2, (short)0, accx, false, false);
        }
    }
    float* out = kv + (size_t)bh * DHEAD * DHEAD;
#pragma unroll
    for (int r = 0; r < 8; ++r)
        out[(i0 * 16 + hi * 8 + r) * DHEAD + j0 * 16 + mrow] = acc[r];
    if (extra) {
#pragma unroll
        for (int r = 0; r < 8; ++r)
            out[(32 + hi * 8 + r) * DHEAD + 32 + mrow] = accx[r];
    }
}

// attn[b][h*48+e][n] = (sum_d q_d * kv[d][e]) / (sum_d q_d*ksum_d + 1e-8)
__global__ void attn_numden_kernel(const float* __restrict__ q, const float* __restrict__ kv,
                                   const float* __restrict__ ks, float* __restrict__ attn, int Ntok) {
    int t = blockIdx.x * 256 + threadIdx.x;
    int b = t / Ntok, n = t % Ntok;
    const float* qb = q + (size_t)b * CEMB * Ntok + n;
    float* ab = attn + (size_t)b * CEMB * Ntok + n;
    for (int h = 0; h < NHEAD; ++h) {
        float qv[DHEAD];
        const float* qh = qb + (size_t)h * DHEAD * Ntok;
#pragma unroll
        for (int d = 0; d < DHEAD; ++d) qv[d] = qh[(size_t)d * Ntok];
        const float* ksh = ks + (b * NHEAD + h) * DHEAD;
        float den = 1e-8f;
#pragma unroll
        for (int d = 0; d < DHEAD; ++d) den += qv[d] * ksh[d];
        float inv = 1.0f / den;
        const float* kvh = kv + (size_t)(b * NHEAD + h) * DHEAD * DHEAD;
        float* ah = ab + (size_t)h * DHEAD * Ntok;
        for (int e = 0; e < DHEAD; ++e) {
            float sum = 0.f;
#pragma unroll
            for (int d = 0; d < DHEAD; ++d) sum += qv[d] * kvh[d * DHEAD + e];
            ah[(size_t)e * Ntok] = sum * inv;
        }
    }
}

// out = base + add * scal[c]
__global__ void residual_kernel(const float* __restrict__ base, const float* __restrict__ add,
                                const float* __restrict__ scal, float* __restrict__ out,
                                int HW, size_t total) {
    size_t i = (size_t)blockIdx.x * 256 + threadIdx.x;
    if (i >= total) return;
    int c = (int)((i / HW) % CEMB);
    out[i] = base[i] + add[i] * scal[c];
}

// out = qimg + (working + ffn*fs[c]) * finals[c]
__global__ void final_kernel(const float* __restrict__ qimg, const float* __restrict__ working,
                             const float* __restrict__ ffn, const float* __restrict__ fs,
                             const float* __restrict__ finals, float* __restrict__ out,
                             int HW, size_t total) {
    size_t i = (size_t)blockIdx.x * 256 + threadIdx.x;
    if (i >= total) return;
    int c = (int)((i / HW) % CEMB);
    float w2 = working[i] + ffn[i] * fs[c];
    out[i] = qimg[i] + w2 * finals[c];
}

// ---------------- main WMMA GEMM ----------------
// OUT[b][f][n] = act( sum_k W[f][k] * (alpha[b][k]*X[b][k][n] + beta[k]) + bias[f] )
// W bf16 row-major MxK; X f32 (or bf16 via Xb) channel-major [B][K][N].
// Tile: 128(M) x 64(N), 8 waves of 32x32 (2x2 WMMA accs), K step 32.
// A tile is DMA'd by the Tensor Data Mover (tensor_load_to_lds) into a
// double-buffered LDS tile while the current step's WMMAs + X-conversion run;
// drained with s_wait_tensorcnt before the single per-iteration barrier.
__global__ __launch_bounds__(256) void gemm_bf16_kernel(
    const unsigned short* __restrict__ Wb,
    const float* __restrict__ Xf,
    const unsigned short* __restrict__ Xb,
    const float* __restrict__ alpha,
    const float* __restrict__ beta,
    const float* __restrict__ bias,
    float* __restrict__ Of,
    unsigned short* __restrict__ Ob,
    int M, int K, int N, int act) {
    __shared__ __align__(16) unsigned short sA[2][128 * 32];   // A tiles, row-major, stride 32
    __shared__ __align__(16) unsigned short sX[2][64 * 40];    // X tiles transposed [n][k], stride 40
    int b = blockIdx.z;
    int n0 = blockIdx.x * 64;
    int f0 = blockIdx.y * 128;
    int tid = threadIdx.x, lane = tid & 31, wave = tid >> 5;
    int wm = wave >> 1, wn = wave & 1;          // wave -> 32x32 subtile
    int mrow = lane & 15, hi = lane >> 4;
    size_t xoff = (size_t)b * K * N;
    size_t ooff = (size_t)b * M * N;

    // ---- A tile producer: TDM DMA (wave 0 issues for whole WG) or manual copy ----
    auto issue_A = [&](int k0, int buf) {
#if HAVE_TDM
        if (wave == 0) {
            unsigned ldsoff = (unsigned)(size_t)(void*)&sA[buf][0];
            unsigned long long ga =
                (unsigned long long)(size_t)(Wb + (size_t)f0 * K + k0);   // byte address of tile start
            // D# group0: count=1 | lds_addr | global_addr | type=2
            u32x4 g0;
            g0[0] = 1u;
            g0[1] = ldsoff;
            g0[2] = (unsigned)(ga & 0xFFFFFFFFu);
            g0[3] = (unsigned)((ga >> 32) & 0x1FFFFFFu) | (2u << 30);
            // D# group1: data_size=2B; tensor dims maxed (tile always in-bounds);
            // tile 32(x) x 128(y); dim0 stride = K elements.
            i32x8 g1;
            g1[0] = (int)(1u << 16);            // data_size = 1 -> 2 bytes
            g1[1] = (int)0xFFFF0000u;           // tensor_dim0 lo16 << 16   (dim0 = 0x7FFFFFFF)
            g1[2] = (int)0xFFFF7FFFu;           // dim0 hi16 | dim1 lo16<<16 (dim1 = 0x7FFFFFFF)
            g1[3] = (int)(0x7FFFu | (32u << 16)); // dim1 hi16 | tile_dim0=32
            g1[4] = 128;                        // tile_dim1=128, tile_dim2=0
            g1[5] = (int)(unsigned)K;           // tensor_dim0_stride lo32
            g1[6] = 0;                          // stride hi16 | dim1_stride lo16
            g1[7] = 0;
            i32x4 gz = {0, 0, 0, 0};
#if __clang_major__ >= 23
            i32x8 gz8 = {0, 0, 0, 0, 0, 0, 0, 0};
            __builtin_amdgcn_tensor_load_to_lds(g0, g1, gz, gz, gz8, 0);
#else
            __builtin_amdgcn_tensor_load_to_lds(g0, g1, gz, gz, 0);
#endif
        }
#else
        {   // fallback: 16 halfs per thread via 2x b128
            int row = tid >> 1, col = (tid & 1) * 16;
            const uint4* src = (const uint4*)(Wb + (size_t)(f0 + row) * K + k0 + col);
            uint4 w0 = src[0], w1 = src[1];
            *((uint4*)&sA[buf][row * 32 + col]) = w0;
            *((uint4*)&sA[buf][row * 32 + col + 8]) = w1;
        }
#endif
    };
    auto wait_A = [&]() {
#if HAVE_TDM
        __builtin_amdgcn_s_wait_tensorcnt(0);   // no-op for waves with TENSORcnt==0
#endif
    };
    // ---- X tile: 32(k) x 64(n), convert + affine + transpose into sX[buf][n][k] ----
    auto stage_X = [&](int k0, int buf) {
        int n = tid & 63;
        int kb = (tid >> 6) * 8;
        if (Xb) {
#pragma unroll
            for (int i = 0; i < 8; ++i) {
                int k = kb + i;
                sX[buf][n * 40 + k] = Xb[xoff + (size_t)(k0 + k) * N + n0 + n];
            }
        } else {
#pragma unroll
            for (int i = 0; i < 8; ++i) {
                int k = kb + i;
                float v = Xf[xoff + (size_t)(k0 + k) * N + n0 + n];
                if (alpha) v *= alpha[b * K + k0 + k];
                if (beta)  v += beta[k0 + k];
                sX[buf][n * 40 + k] = f2bf(v);
            }
        }
    };

    floatx8 acc[2][2] = {};

    // pipelined main loop: prefetch next A (TDM) + next X while computing current
    issue_A(0, 0);
    stage_X(0, 0);
    wait_A();
    __syncthreads();

    int nsteps = K / 32;
    for (int s = 0; s < nsteps; ++s) {
        int cur = s & 1, nxt = cur ^ 1;
        bool more = (s + 1) < nsteps;
        if (more) issue_A((s + 1) * 32, nxt);

        bf16x16 afr[2], bfr[2];
        afr[0] = load_a_frag(&sA[cur][(wm * 32 + mrow) * 32], hi);
        afr[1] = load_a_frag(&sA[cur][(wm * 32 + 16 + mrow) * 32], hi);
        bfr[0] = load_b_frag(&sX[cur][(wn * 32 + mrow) * 40], hi);
        bfr[1] = load_b_frag(&sX[cur][(wn * 32 + 16 + mrow) * 40], hi);
#pragma unroll
        for (int mt = 0; mt < 2; ++mt)
#pragma unroll
            for (int nt = 0; nt < 2; ++nt)
                acc[mt][nt] = __builtin_amdgcn_wmma_f32_16x16x32_bf16(
                    false, afr[mt], false, bfr[nt], (short)0, acc[mt][nt], false, false);

        if (more) stage_X((s + 1) * 32, nxt);
        wait_A();
        __syncthreads();
    }

    // epilogue: C layout -> f = tile_f0 + hi*8 + r, n = tile_n0 + (lane&15)
#pragma unroll
    for (int mt = 0; mt < 2; ++mt)
#pragma unroll
        for (int nt = 0; nt < 2; ++nt)
#pragma unroll
            for (int r = 0; r < 8; ++r) {
                int f = f0 + wm * 32 + mt * 16 + hi * 8 + r;
                int n = n0 + wn * 32 + nt * 16 + mrow;
                float v = acc[mt][nt][r];
                if (bias) v += bias[f];
                if (act == 1) v = v / (1.0f + __expf(-v));   // silu
                if (Ob) Ob[ooff + (size_t)f * N + n] = f2bf(v);
                else    Of[ooff + (size_t)f * N + n] = v;
            }
}

// ---------------- host launcher ----------------
extern "C" void kernel_launch(void* const* d_in, const int* in_sizes, int n_in,
                              void* d_out, int out_size, void* d_ws, size_t ws_size,
                              hipStream_t stream) {
    const float* qimg  = (const float*)d_in[0];
    const float* kimg  = (const float*)d_in[1];
    const float* vimg  = (const float*)d_in[2];
    const float* adag  = (const float*)d_in[3];
    const float* adab  = (const float*)d_in[4];
    const float* Wq    = (const float*)d_in[5];
    const float* bq    = (const float*)d_in[6];
    const float* Wk    = (const float*)d_in[7];
    const float* bk    = (const float*)d_in[8];
    const float* Wv    = (const float*)d_in[9];
    const float* bv    = (const float*)d_in[10];
    const float* Wo    = (const float*)d_in[11];
    const float* bo    = (const float*)d_in[12];
    const float* lnqw  = (const float*)d_in[13];
    const float* lnqb  = (const float*)d_in[14];
    const float* lnkw  = (const float*)d_in[15];
    const float* lnkb  = (const float*)d_in[16];
    const float* ascal = (const float*)d_in[17];
    const float* ffng  = (const float*)d_in[18];
    const float* ffnb  = (const float*)d_in[19];
    const float* W1    = (const float*)d_in[20];
    const float* b1    = (const float*)d_in[21];
    const float* W2    = (const float*)d_in[22];
    const float* b2    = (const float*)d_in[23];
    const float* fscal = (const float*)d_in[24];
    const float* finsc = (const float*)d_in[25];
    float* out = (float*)d_out;

    const int B = in_sizes[0] / (CEMB * NQTOK);

    // workspace carve-up
    char* ws = (char*)d_ws;
    size_t off = 0;
    auto alloc = [&](size_t bytes) -> void* {
        off = (off + 255) & ~(size_t)255;
        void* p = ws + off;
        off += bytes;
        return p;
    };
    unsigned short* wWq = (unsigned short*)alloc((size_t)CEMB * CEMB * 2);
    unsigned short* wWk = (unsigned short*)alloc((size_t)CEMB * CEMB * 2);
    unsigned short* wWv = (unsigned short*)alloc((size_t)CEMB * CEMB * 2);
    unsigned short* wWo = (unsigned short*)alloc((size_t)CEMB * CEMB * 2);
    unsigned short* wW1 = (unsigned short*)alloc((size_t)FFDIM * CEMB * 2);
    unsigned short* wW2 = (unsigned short*)alloc((size_t)CEMB * FFDIM * 2);
    float* gx1    = (float*)alloc((size_t)B * CEMB * 4);
    float* alpha1 = (float*)alloc((size_t)B * CEMB * 4);
    float* gx2    = (float*)alloc((size_t)B * CEMB * 4);
    float* alpha2 = (float*)alloc((size_t)B * CEMB * 4);
    float* ksum   = (float*)alloc((size_t)B * CEMB * 4);
    float* kvbuf  = (float*)alloc((size_t)B * NHEAD * DHEAD * DHEAD * 4);
    float* q_pre  = (float*)alloc((size_t)B * CEMB * NQTOK * 4);   // reused as attn
    float* q_post = (float*)alloc((size_t)B * CEMB * NQTOK * 4);
    float* k_pre  = (float*)alloc((size_t)B * CEMB * NKTOK * 4);
    float* k_post = (float*)alloc((size_t)B * CEMB * NKTOK * 4);
    float* vbuf   = (float*)alloc((size_t)B * CEMB * NKTOK * 4);
    float* a_out  = (float*)alloc((size_t)B * CEMB * NQTOK * 4);
    float* workb  = (float*)alloc((size_t)B * CEMB * NQTOK * 4);
    unsigned short* hbuf = (unsigned short*)alloc((size_t)B * FFDIM * NQTOK * 2);
    float* ffout  = (float*)alloc((size_t)B * CEMB * NQTOK * 4);
    float* attn   = q_pre;   // alias: q_pre dead after ln_elu(q)
    (void)ws_size;

    // 1. weights -> bf16
    auto cvt = [&](const float* s, unsigned short* d, int n) {
        f32_to_bf16_kernel<<<(n + 255) / 256, 256, 0, stream>>>(s, d, n);
    };
    cvt(Wq, wWq, CEMB * CEMB); cvt(Wk, wWk, CEMB * CEMB); cvt(Wv, wWv, CEMB * CEMB);
    cvt(Wo, wWo, CEMB * CEMB); cvt(W1, wW1, FFDIM * CEMB); cvt(W2, wW2, CEMB * FFDIM);

    // 2. GRN #1 -> per-(b,c) affine
    grn_reduce_kernel<<<B * CEMB, 256, 0, stream>>>(qimg, gx1, NQTOK);
    grn_final_kernel<<<B, CEMB, 0, stream>>>(gx1, adag, alpha1);

    auto gemm = [&](const unsigned short* Wb_, const float* Xf_, const unsigned short* Xb_,
                    const float* al_, const float* be_, const float* bi_,
                    float* Of_, unsigned short* Ob_, int M, int K, int N, int act) {
        dim3 grid(N / 64, M / 128, B);
        gemm_bf16_kernel<<<grid, 256, 0, stream>>>(Wb_, Xf_, Xb_, al_, be_, bi_, Of_, Ob_, M, K, N, act);
    };

    // 3. projections (GRN affine folded into Q's input)
    gemm(wWq, qimg, nullptr, alpha1, adab, bq, q_pre, nullptr, CEMB, CEMB, NQTOK, 0);
    gemm(wWk, kimg, nullptr, nullptr, nullptr, bk, k_pre, nullptr, CEMB, CEMB, NKTOK, 0);
    gemm(wWv, vimg, nullptr, nullptr, nullptr, bv, vbuf, nullptr, CEMB, CEMB, NKTOK, 0);

    // 4. LN + elu+1
    ln_elu_kernel<<<(B * NQTOK) / 256, 256, 0, stream>>>(q_pre, lnqw, lnqb, q_post, NQTOK);
    ln_elu_kernel<<<(B * NKTOK) / 256, 256, 0, stream>>>(k_pre, lnkw, lnkb, k_post, NKTOK);

    // 5. linear-attention state
    ksum_kernel<<<B * CEMB, 256, 0, stream>>>(k_post, ksum, NKTOK);
    kv_outer_kernel<<<B * NHEAD, 256, 0, stream>>>(k_post, vbuf, kvbuf);
    attn_numden_kernel<<<(B * NQTOK) / 256, 256, 0, stream>>>(q_post, kvbuf, ksum, attn, NQTOK);

    // 6. output projection + residual
    gemm(wWo, attn, nullptr, nullptr, nullptr, bo, a_out, nullptr, CEMB, CEMB, NQTOK, 0);
    {
        size_t total = (size_t)B * CEMB * NQTOK;
        residual_kernel<<<(unsigned)((total + 255) / 256), 256, 0, stream>>>(qimg, a_out, ascal, workb, NQTOK, total);
    }

    // 7. GRN #2 -> affine, FFN (W1+silu -> bf16 h, then W2)
    grn_reduce_kernel<<<B * CEMB, 256, 0, stream>>>(workb, gx2, NQTOK);
    grn_final_kernel<<<B, CEMB, 0, stream>>>(gx2, ffng, alpha2);
    gemm(wW1, workb, nullptr, alpha2, ffnb, b1, nullptr, hbuf, FFDIM, CEMB, NQTOK, 1);
    gemm(wW2, nullptr, hbuf, nullptr, nullptr, b2, ffout, nullptr, CEMB, FFDIM, NQTOK, 0);

    // 8. final residual chain
    {
        size_t total = (size_t)B * CEMB * NQTOK;
        final_kernel<<<(unsigned)((total + 255) / 256), 256, 0, stream>>>(qimg, workb, ffout, fscal, finsc, out, NQTOK, total);
    }
}
